// Attention_20830591385623
// MI455X (gfx1250) — compile-verified, matrix operand order checked by reading, not run on hardware
//
#include <hip/hip_runtime.h>
#include <hip/hip_bf16.h>

typedef __attribute__((ext_vector_type(16))) _Float16 v16h;
typedef __attribute__((ext_vector_type(8)))  float    v8f;
typedef __attribute__((ext_vector_type(4)))  int      v4i;

#define D_MODEL 4544
#define E_FUSED 4672
#define NHEAD   71
#define DKV     64
#define SEQ     1024
#define NB      2

#define AS1 __attribute__((address_space(1)))
#define AS3 __attribute__((address_space(3)))

#if defined(__has_builtin)
#if __has_builtin(__builtin_amdgcn_global_load_async_to_lds_b128)
#define HAVE_ASYNC_LDS 1
#endif
#endif

// ---------------------------------------------------------------------------
// helpers
// ---------------------------------------------------------------------------
__device__ inline v8f zero8() {
  v8f z;
#pragma unroll
  for (int i = 0; i < 8; ++i) z[i] = 0.0f;
  return z;
}

// Build a 16-bit WMMA A/B fragment (16x32 / 32x16) from a row of 32 f16
// values.  ISA layout: lanes 0-15 hold K = 0..7 (vgpr0-3) and 16..23
// (vgpr4-7); lanes 16-31 hold K = 8..15 and 24..31.
__device__ inline v16h load_frag16(const _Float16* row, int half) {
  v16h f;
  const _Float16* p0 = row + half * 8;
  const _Float16* p1 = row + 16 + half * 8;
#pragma unroll
  for (int e = 0; e < 8; ++e) { f[e] = p0[e]; f[8 + e] = p1[e]; }
  return f;
}

// 16-byte global -> LDS copy.  Async DMA path on CDNA5 (ASYNCcnt-tracked,
// overlaps with WMMA), synchronous fallback otherwise.
__device__ inline void copy16_g2l(_Float16* lds_dst, const _Float16* gsrc) {
#if defined(HAVE_ASYNC_LDS)
  __builtin_amdgcn_global_load_async_to_lds_b128(
      (AS1 v4i*)(uintptr_t)gsrc,
      (AS3 v4i*)(unsigned)(uintptr_t)lds_dst, 0, 0);
#else
  *(uint4*)lds_dst = *(const uint4*)gsrc;
#endif
}

__device__ inline void wait_g2l() {
#if defined(HAVE_ASYNC_LDS)
#if __has_builtin(__builtin_amdgcn_s_wait_asynccnt)
  __builtin_amdgcn_s_wait_asynccnt(0);
#else
  asm volatile("s_wait_asynccnt 0x0" ::: "memory");
#endif
#endif
}

// ---------------------------------------------------------------------------
// fp32 -> f16 conversion
// ---------------------------------------------------------------------------
__global__ __launch_bounds__(256) void cvt_f32_f16(const float* __restrict__ in,
                                                   _Float16* __restrict__ out,
                                                   int n) {
  int i = blockIdx.x * 256 + threadIdx.x;
  if (i < n) out[i] = (_Float16)in[i];
}

// ---------------------------------------------------------------------------
// C(MxN, f32 or f16) = A(MxK, f16 row-major) * B(NxK, f16 row-major)^T
// BM=256 BN=64 BK=32, 256 threads = 8 waves (4x2), wave tile 64x32.
// Double-buffered LDS with async global->LDS staging.
// ---------------------------------------------------------------------------
#define GBM 256
#define GBN 64

template <bool OUT_F16>
__global__ __launch_bounds__(256) void gemm_abT(const _Float16* __restrict__ A,
                                                const _Float16* __restrict__ B,
                                                void* __restrict__ Cout,
                                                int M, int N, int K) {
  __shared__ _Float16 As[2][GBM][40];   // +8 f16 pad -> conflict free
  __shared__ _Float16 Bs[2][GBN][40];

  const int tid  = threadIdx.x;
  const int lane = tid & 31;
  const int wave = tid >> 5;
  const int half = lane >> 4;
  const int ln   = lane & 15;
  const int waveM = wave >> 1;       // 0..3  -> 64 rows each
  const int waveN = wave & 1;        // 0..1  -> 32 cols each
  const int m0 = blockIdx.y * GBM;
  const int n0 = blockIdx.x * GBN;

  v8f acc[4][2];
#pragma unroll
  for (int i = 0; i < 4; ++i)
#pragma unroll
    for (int j = 0; j < 2; ++j) acc[i][j] = zero8();

  // tile staging: A = 256x32 (1024 16B chunks), B = 64x32 (256 chunks)
  auto stage = [&](int buf, int kk) {
#pragma unroll
    for (int c = tid; c < 1024; c += 256) {
      const int row = c >> 2, kc = (c & 3) * 8;
      copy16_g2l(&As[buf][row][kc], A + (size_t)(m0 + row) * K + kk + kc);
    }
    {
      const int row = tid >> 2, kc = (tid & 3) * 8;
      copy16_g2l(&Bs[buf][row][kc], B + (size_t)(n0 + row) * K + kk + kc);
    }
  };

  const int kTiles = K / 32;
  stage(0, 0);
  wait_g2l();
  __syncthreads();

  for (int kb = 0; kb < kTiles; ++kb) {
    const int cur = kb & 1;
    if (kb + 1 < kTiles) stage(cur ^ 1, (kb + 1) * 32);   // overlap with WMMA

    v16h af[4], bf[2];
#pragma unroll
    for (int mt = 0; mt < 4; ++mt)
      af[mt] = load_frag16(&As[cur][waveM * 64 + mt * 16 + ln][0], half);
#pragma unroll
    for (int nt = 0; nt < 2; ++nt)
      bf[nt] = load_frag16(&Bs[cur][waveN * 32 + nt * 16 + ln][0], half);

#pragma unroll
    for (int mt = 0; mt < 4; ++mt)
#pragma unroll
      for (int nt = 0; nt < 2; ++nt)
        acc[mt][nt] = __builtin_amdgcn_wmma_f32_16x16x32_f16(
            false, af[mt], false, bf[nt], (short)0, acc[mt][nt], false, false);

    wait_g2l();          // this wave's next-tile DMA done
    __syncthreads();     // everyone's staging visible; cur safe to recycle
  }

  // epilogue: C layout -> lane (half, ln) holds rows r+8*half, col ln
#pragma unroll
  for (int mt = 0; mt < 4; ++mt)
#pragma unroll
    for (int nt = 0; nt < 2; ++nt) {
      const int gcol = n0 + waveN * 32 + nt * 16 + ln;
#pragma unroll
      for (int r = 0; r < 8; ++r) {
        const int grow = m0 + waveM * 64 + mt * 16 + r + 8 * half;
        const float v = acc[mt][nt][r];
        if (OUT_F16)
          ((_Float16*)Cout)[(size_t)grow * N + gcol] = (_Float16)v;
        else
          ((float*)Cout)[(size_t)grow * N + gcol] = v;
      }
    }
}

// ---------------------------------------------------------------------------
// RoPE + head split.  fused: [NB*SEQ][4672] f16.
// Q out: [n][h][l][64]; K out: [n][l][64].  One thread per (n,l,head,d<32).
// ---------------------------------------------------------------------------
__global__ __launch_bounds__(256) void rope_kernel(const _Float16* __restrict__ fused,
                                                   _Float16* __restrict__ Q,
                                                   _Float16* __restrict__ Kc) {
  const int idx  = blockIdx.x * 256 + threadIdx.x;   // < NB*SEQ*72*32
  const int d    = idx & 31;
  const int head = (idx >> 5) % 72;
  const int t    = (idx >> 5) / 72;                  // n*SEQ + l
  const int l    = t % SEQ;
  const int nn   = t / SEQ;
  const size_t frow = (size_t)t * E_FUSED;
  const int col = (head < NHEAD) ? head * 64 : D_MODEL;   // key head at 4544
  const float x0 = (float)fused[frow + col + d];
  const float x1 = (float)fused[frow + col + d + 32];
  // inv_freq = 10000^(-2d/64)
  const float f = __expf(-(float)d * (2.0f / 64.0f) * 9.210340371976184f);
  float sn, cs;
  __sincosf((float)l * f, &sn, &cs);
  const float y0 = x0 * cs - x1 * sn;
  const float y1 = x1 * cs + x0 * sn;
  if (head < NHEAD) {
    const size_t qo = (((size_t)nn * NHEAD + head) * SEQ + l) * 64;
    Q[qo + d]      = (_Float16)y0;
    Q[qo + d + 32] = (_Float16)y1;
  } else {
    const size_t ko = ((size_t)nn * SEQ + l) * 64;
    Kc[ko + d]      = (_Float16)y0;
    Kc[ko + d + 32] = (_Float16)y1;
  }
}

// V transposed: VT[n][d][l] so B-fragment rows (columns of V) are contiguous.
__global__ __launch_bounds__(256) void vtrans_kernel(const _Float16* __restrict__ fused,
                                                     _Float16* __restrict__ VT) {
  const int idx = blockIdx.x * 256 + threadIdx.x;    // < NB*SEQ*64
  const int d = idx & 63;
  const int t = idx >> 6;
  const int l = t % SEQ;
  const int nn = t / SEQ;
  VT[((size_t)nn * 64 + d) * SEQ + l] =
      fused[(size_t)t * E_FUSED + D_MODEL + DKV + d];
}

// ---------------------------------------------------------------------------
// Flash attention, causal, GQA (1 KV head).  Grid (SEQ/64, NHEAD, NB),
// 128 threads = 4 waves, each wave owns 16 query rows.
// ---------------------------------------------------------------------------
__global__ __launch_bounds__(128) void attn_kernel(const _Float16* __restrict__ Q,
                                                   const _Float16* __restrict__ Kc,
                                                   const _Float16* __restrict__ VT,
                                                   _Float16* __restrict__ Ao) {
  __shared__ _Float16 Pl[4][16][40];   // per-wave P staging (C->A relayout)
  const int qb = blockIdx.x, h = blockIdx.y, nn = blockIdx.z;
  const int tid = threadIdx.x, lane = tid & 31, w = tid >> 5;
  const int half = lane >> 4, ln = lane & 15;
  const _Float16* qbase = Q + ((size_t)nn * NHEAD + h) * SEQ * DKV;
  const _Float16* kbase = Kc + (size_t)nn * SEQ * DKV;
  const _Float16* vbase = VT + (size_t)nn * DKV * SEQ;
  const int qr0 = qb * 64 + w * 16;

  v16h qf[2];
#pragma unroll
  for (int c = 0; c < 2; ++c)
    qf[c] = load_frag16(qbase + (size_t)(qr0 + ln) * DKV + c * 32, half);

  v8f o[4];
#pragma unroll
  for (int nc = 0; nc < 4; ++nc) o[nc] = zero8();
  float mrow[8], lrow[8];
#pragma unroll
  for (int r = 0; r < 8; ++r) { mrow[r] = -3.0e38f; lrow[r] = 0.0f; }

  const int jmax = 2 * qb + 2;          // kv blocks of 32 covering [0, qb*64+64)
  for (int j = 0; j < jmax; ++j) {
    const int kv0 = j * 32;
    // S(16x32) = Q(16x64) @ K^T  (two k-chunks chained into each 16x16 tile)
    v8f s[2];
#pragma unroll
    for (int nt = 0; nt < 2; ++nt) {
      v8f t = zero8();
#pragma unroll
      for (int c = 0; c < 2; ++c) {
        v16h bf = load_frag16(kbase + (size_t)(kv0 + nt * 16 + ln) * DKV + c * 32, half);
        t = __builtin_amdgcn_wmma_f32_16x16x32_f16(false, qf[c], false, bf,
                                                   (short)0, t, false, false);
      }
      s[nt] = t;
    }
    // scale + causal mask + online softmax
    float p[2][8], mx[8], rs[8], alpha[8];
#pragma unroll
    for (int r = 0; r < 8; ++r) {
      const int rabs = qr0 + r + 8 * half;
      float s0 = s[0][r] * 0.125f; if (kv0 + ln > rabs)      s0 = -3.0e38f;
      float s1 = s[1][r] * 0.125f; if (kv0 + 16 + ln > rabs) s1 = -3.0e38f;
      p[0][r] = s0; p[1][r] = s1;
      mx[r] = fmaxf(s0, s1);
    }
#pragma unroll
    for (int off = 1; off < 16; off <<= 1)
#pragma unroll
      for (int r = 0; r < 8; ++r) mx[r] = fmaxf(mx[r], __shfl_xor(mx[r], off, 32));
#pragma unroll
    for (int r = 0; r < 8; ++r) {
      const float mn = fmaxf(mrow[r], mx[r]);
      alpha[r] = __expf(mrow[r] - mn);
      mrow[r] = mn;
      const float p0 = __expf(p[0][r] - mn);
      const float p1 = __expf(p[1][r] - mn);
      p[0][r] = p0; p[1][r] = p1;
      rs[r] = p0 + p1;
    }
#pragma unroll
    for (int off = 1; off < 16; off <<= 1)
#pragma unroll
      for (int r = 0; r < 8; ++r) rs[r] += __shfl_xor(rs[r], off, 32);
#pragma unroll
    for (int r = 0; r < 8; ++r) lrow[r] = lrow[r] * alpha[r] + rs[r];

    // relayout P: C-layout (lane=col, vgpr=row) -> A-layout via per-wave LDS
#pragma unroll
    for (int r = 0; r < 8; ++r) {
      Pl[w][r + 8 * half][ln]      = (_Float16)p[0][r];
      Pl[w][r + 8 * half][16 + ln] = (_Float16)p[1][r];
    }
    asm volatile("s_wait_dscnt 0x0" ::: "memory");
    v16h pf = load_frag16(&Pl[w][ln][0], half);

    // O = alpha*O + P @ V  (V^T rows are contiguous per output column)
#pragma unroll
    for (int nc = 0; nc < 4; ++nc) {
#pragma unroll
      for (int r = 0; r < 8; ++r) o[nc][r] *= alpha[r];
      v16h vf = load_frag16(vbase + (size_t)(nc * 16 + ln) * SEQ + kv0, half);
      o[nc] = __builtin_amdgcn_wmma_f32_16x16x32_f16(false, pf, false, vf,
                                                     (short)0, o[nc], false, false);
    }
    asm volatile("s_wait_dscnt 0x0" ::: "memory");
  }

  // normalize and write [n][l][h*64 + d]
#pragma unroll
  for (int nc = 0; nc < 4; ++nc)
#pragma unroll
    for (int r = 0; r < 8; ++r) {
      const int li = qr0 + r + 8 * half;
      const float val = o[nc][r] / lrow[r];
      Ao[((size_t)nn * SEQ + li) * D_MODEL + h * DKV + nc * 16 + ln] = (_Float16)val;
    }
}

// ---------------------------------------------------------------------------
// host
// ---------------------------------------------------------------------------
extern "C" void kernel_launch(void* const* d_in, const int* in_sizes, int n_in,
                              void* d_out, int out_size, void* d_ws, size_t ws_size,
                              hipStream_t stream) {
  const float* hs     = (const float*)d_in[0];
  const float* wqkv   = (const float*)d_in[1];
  const float* wdense = (const float*)d_in[2];
  float* out = (float*)d_out;

  const size_t nHS = (size_t)NB * SEQ * D_MODEL;    // 9,306,112
  const size_t nWQ = (size_t)E_FUSED * D_MODEL;     // 21,229,568
  const size_t nWD = (size_t)D_MODEL * D_MODEL;     // 20,647,936
  const size_t nFU = (size_t)NB * SEQ * E_FUSED;    // 9,568,256
  const size_t nQ  = (size_t)NB * NHEAD * SEQ * DKV;
  const size_t nK  = (size_t)NB * SEQ * DKV;

  char* ws = (char*)d_ws;
  _Float16* hsf   = (_Float16*)ws; ws += nHS * 2;
  _Float16* wqf   = (_Float16*)ws; ws += nWQ * 2;
  _Float16* wdf   = (_Float16*)ws; ws += nWD * 2;
  _Float16* fused = (_Float16*)ws; ws += nFU * 2;
  _Float16* Qb    = (_Float16*)ws; ws += nQ * 2;
  _Float16* Kb    = (_Float16*)ws; ws += nK * 2;
  _Float16* VTb   = (_Float16*)ws; ws += nK * 2;
  _Float16* Ab    = (_Float16*)ws; ws += nHS * 2;

  cvt_f32_f16<<<(int)((nHS + 255) / 256), 256, 0, stream>>>(hs, hsf, (int)nHS);
  cvt_f32_f16<<<(int)((nWQ + 255) / 256), 256, 0, stream>>>(wqkv, wqf, (int)nWQ);
  cvt_f32_f16<<<(int)((nWD + 255) / 256), 256, 0, stream>>>(wdense, wdf, (int)nWD);

  gemm_abT<true><<<dim3(E_FUSED / GBN, (NB * SEQ) / GBM), 256, 0, stream>>>(
      hsf, wqf, (void*)fused, NB * SEQ, E_FUSED, D_MODEL);

  rope_kernel<<<(NB * SEQ * 72 * 32) / 256, 256, 0, stream>>>(fused, Qb, Kb);
  vtrans_kernel<<<(NB * SEQ * 64) / 256, 256, 0, stream>>>(fused, VTb);

  attn_kernel<<<dim3(SEQ / 64, NHEAD, NB), 128, 0, stream>>>(Qb, Kb, VTb, Ab);

  gemm_abT<false><<<dim3(D_MODEL / GBN, (NB * SEQ) / GBM), 256, 0, stream>>>(
      Ab, wdf, (void*)out, NB * SEQ, D_MODEL, D_MODEL);
}
